// GATProcessor_19121194401846
// MI455X (gfx1250) — matrix operand order repaired and checked by reading, not compiled, not run on hardware
//
#include <hip/hip_runtime.h>
#include <math.h>

#define N_NODES 8192
#define C_DIM   256
#define KNN_K   16
#define NEG_SLOPE 0.2f

typedef __attribute__((ext_vector_type(2)))  float  v2f;
typedef __attribute__((ext_vector_type(8)))  float  v8f;
typedef __attribute__((ext_vector_type(16))) __bf16 v16bf;
typedef __attribute__((ext_vector_type(8)))  __bf16 v8bf;

union V16U { v16bf v; v8bf h[2]; };

__device__ __forceinline__ float wave_sum(float v) {
#pragma unroll
  for (int m = 16; m >= 1; m >>= 1) v += __shfl_xor(v, m, 32);
  return v;
}
__device__ __forceinline__ float wave_max(float v) {
#pragma unroll
  for (int m = 16; m >= 1; m >>= 1) v = fmaxf(v, __shfl_xor(v, m, 32));
  return v;
}

__device__ __forceinline__ unsigned short f2bf_rne(float f) {
  unsigned u = __float_as_uint(f);
  return (unsigned short)((u + 0x7FFFu + ((u >> 16) & 1u)) >> 16);
}
__device__ __forceinline__ float bf2f(unsigned short b) {
  return __uint_as_float((unsigned)b << 16);
}

// -------------------- sq[i] = ||x_i||^2 --------------------
__global__ void rowsq_kernel(const float* __restrict__ x, float* __restrict__ sq) {
  const int wave = threadIdx.x >> 5, lane = threadIdx.x & 31;
  const int node = blockIdx.x * 8 + wave;
  const float* xr = x + (size_t)node * C_DIM;
  float acc = 0.f;
#pragma unroll
  for (int t = 0; t < C_DIM / 32; ++t) { float v = xr[lane + 32 * t]; acc += v * v; }
  acc = wave_sum(acc);
  if (lane == 0) sq[node] = acc;
}

// -------------------- x -> (hi, lo) bf16 split (RNE) --------------------
__global__ void cvt_hilo_kernel(const float* __restrict__ x,
                                unsigned short* __restrict__ hi,
                                unsigned short* __restrict__ lo) {
  const size_t i = ((size_t)blockIdx.x * blockDim.x + threadIdx.x) * 4;
  const float4 v = *(const float4*)(x + i);
  ushort4 h, l;
  h.x = f2bf_rne(v.x); l.x = f2bf_rne(v.x - bf2f(h.x));
  h.y = f2bf_rne(v.y); l.y = f2bf_rne(v.y - bf2f(h.y));
  h.z = f2bf_rne(v.z); l.z = f2bf_rne(v.z - bf2f(h.z));
  h.w = f2bf_rne(v.w); l.w = f2bf_rne(v.w - bf2f(h.w));
  *(ushort4*)(hi + i) = h;
  *(ushort4*)(lo + i) = l;
}

// -------------------- fused x@x.T (split-bf16 WMMA 16x16x32) + running top-k ------
// One wave per 16-row block; A (hi+lo) register-resident for the whole sweep.
// dot ~= hi*hi + hi*lo + lo*hi  (lo*lo term ~2^-16 relative, dropped)
__global__ void __launch_bounds__(32)
knn_kernel(const __bf16* __restrict__ xhi, const __bf16* __restrict__ xlo,
           const float* __restrict__ sq, int* __restrict__ idx_out) {
  __shared__ float Ct[16][16];
  __shared__ float bestd[16][KNN_K];
  __shared__ int   besti[16][KNN_K];

  const int lane = threadIdx.x;
  const int rb   = blockIdx.x * 16;
  const int nloc = lane & 15;          // tile column; also A row index
  const int hh   = lane >> 4;          // half select

  // ---- preload A fragments (hi & lo) into registers: 8 chunks of K=32 ----
  // 16-bit A layout: low lanes K=c*32+0..15, high lanes K=c*32+16..31 (contiguous)
  v16bf Ahi[C_DIM / 32], Alo[C_DIM / 32];
  {
    const __bf16* rh = xhi + (size_t)(rb + nloc) * C_DIM;
    const __bf16* rl = xlo + (size_t)(rb + nloc) * C_DIM;
#pragma unroll
    for (int c = 0; c < C_DIM / 32; ++c) {
      const int k0 = c * 32;
      V16U a;
      a.h[0] = *(const v8bf*)(rh + k0 + 16 * hh);
      a.h[1] = *(const v8bf*)(rh + k0 + 16 * hh + 8);
      Ahi[c] = a.v;
      a.h[0] = *(const v8bf*)(rl + k0 + 16 * hh);
      a.h[1] = *(const v8bf*)(rl + k0 + 16 * hh + 8);
      Alo[c] = a.v;
    }
  }
  if (lane < 16) {
#pragma unroll
    for (int t = 0; t < KNN_K; ++t) { bestd[lane][t] = INFINITY; besti[lane][t] = -1; }
  }
  __syncthreads();

  float bmax = INFINITY; int bpos = 0;   // running worst-of-best (lanes 0..15)

  for (int jb = 0; jb < N_NODES; jb += 16) {
    const int col = jb + nloc;
    const float sqc = sq[col];
    const __bf16* ch = xhi + (size_t)col * C_DIM;
    const __bf16* cl = xlo + (size_t)col * C_DIM;

    v8f acc = {0.f, 0.f, 0.f, 0.f, 0.f, 0.f, 0.f, 0.f};
#pragma unroll
    for (int c = 0; c < C_DIM / 32; ++c) {
      const int k0 = c * 32;
      // 16-bit B layout: low lanes K=k0+0..15, high lanes K=k0+16..31 (contiguous)
      const v16bf bhi = *(const v16bf*)(ch + k0 + 16 * hh);
      const v16bf blo = *(const v16bf*)(cl + k0 + 16 * hh);
      acc = __builtin_amdgcn_wmma_f32_16x16x32_bf16(false, Ahi[c], false, bhi,
                                                    (short)0, acc, false, false);
      acc = __builtin_amdgcn_wmma_f32_16x16x32_bf16(false, Ahi[c], false, blo,
                                                    (short)0, acc, false, false);
      acc = __builtin_amdgcn_wmma_f32_16x16x32_bf16(false, Alo[c], false, bhi,
                                                    (short)0, acc, false, false);
    }

    // ranking key: sq[col] - 2*dot (sq[row] constant within row -> dropped)
#pragma unroll
    for (int r = 0; r < 8; ++r) {
      const int m = r + 8 * hh;
      float key = sqc - 2.0f * acc[r];
      if (rb + m == col) key = INFINITY;      // exclude self (loop=False)
      Ct[m][nloc] = key;
    }
    __syncthreads();

    if (lane < 16) {                          // lane owns row `lane`
#pragma unroll 1
      for (int t = 0; t < 16; ++t) {
        const float d = Ct[lane][t];
        if (d < bmax) {
          bestd[lane][bpos] = d; besti[lane][bpos] = jb + t;
          bmax = -INFINITY;
#pragma unroll
          for (int u = 0; u < KNN_K; ++u) {
            const float bu = bestd[lane][u];
            if (bu > bmax) { bmax = bu; bpos = u; }
          }
        }
      }
    }
    __syncthreads();
  }

  if (lane < 16) {
    const int grow = rb + lane;
#pragma unroll
    for (int t = 0; t < KNN_K; ++t) idx_out[grow * (KNN_K + 1) + t] = besti[lane][t];
    idx_out[grow * (KNN_K + 1) + KNN_K] = grow;   // self loop
  }
}

// -------------------- h = x @ W  (exact fp32 WMMA 16x16x4) --------------------
__global__ void __launch_bounds__(32)
gemm_h_kernel(const float* __restrict__ x, const float* __restrict__ W,
              float* __restrict__ h) {
  __shared__ float Ab[16][C_DIM + 2];
  const int lane = threadIdx.x;
  const int rb = blockIdx.x * 16;
  const int cb = blockIdx.y * 16;

  for (int f = lane * 4; f < 16 * C_DIM; f += 32 * 4) {
    const int r = f >> 8, c = f & (C_DIM - 1);
    const float4 v = *(const float4*)(x + (size_t)(rb + r) * C_DIM + c);
    Ab[r][c + 0] = v.x; Ab[r][c + 1] = v.y; Ab[r][c + 2] = v.z; Ab[r][c + 3] = v.w;
  }
  __syncthreads();

  const int nloc = lane & 15, hh = lane >> 4;
  const float* wc = W + cb + nloc;
  v8f acc = {0.f, 0.f, 0.f, 0.f, 0.f, 0.f, 0.f, 0.f};
#pragma unroll 8
  for (int k0 = 0; k0 < C_DIM; k0 += 4) {
    v2f a = *(const v2f*)&Ab[nloc][k0 + 2 * hh];
    v2f b;
    b.x = wc[(size_t)(k0 + 2 * hh) * C_DIM];
    b.y = wc[(size_t)(k0 + 2 * hh + 1) * C_DIM];
    acc = __builtin_amdgcn_wmma_f32_16x16x4_f32(false, a, false, b,
                                                (short)0, acc, false, false);
  }
#pragma unroll
  for (int r = 0; r < 8; ++r) {
    const int m = r + 8 * hh;
    h[(size_t)(rb + m) * C_DIM + cb + nloc] = acc[r];
  }
}

// -------------------- alpha_src/dst = h @ a_s, h @ a_d --------------------
__global__ void alpha_kernel(const float* __restrict__ h, const float* __restrict__ a_s,
                             const float* __restrict__ a_d, float* __restrict__ as_out,
                             float* __restrict__ ad_out) {
  const int wave = threadIdx.x >> 5, lane = threadIdx.x & 31;
  const int node = blockIdx.x * 8 + wave;
  const float* hr = h + (size_t)node * C_DIM;
  float s = 0.f, d = 0.f;
#pragma unroll
  for (int t = 0; t < C_DIM / 32; ++t) {
    const int c = lane + 32 * t;
    const float v = hr[c];
    s += v * a_s[c]; d += v * a_d[c];
  }
  s = wave_sum(s); d = wave_sum(d);
  if (lane == 0) { as_out[node] = s; ad_out[node] = d; }
}

// -------------------- softmax attention + aggregation + residual ReLU ------------
__global__ void aggregate_kernel(const float* __restrict__ x, const float* __restrict__ h,
                                 const int* __restrict__ idx, const float* __restrict__ asrc,
                                 const float* __restrict__ adst, const float* __restrict__ bias,
                                 float* __restrict__ xout) {
  __shared__ float att_s[8][KNN_K + 1];
  __shared__ int   nb_s[8][KNN_K + 1];
  const int wave = threadIdx.x >> 5, lane = threadIdx.x & 31;
  const int node = blockIdx.x * 8 + wave;

  float e = -INFINITY; int nb = 0;
  if (lane <= KNN_K) {
    nb = idx[node * (KNN_K + 1) + lane];
    const float v = asrc[nb] + adst[node];
    e = v > 0.f ? v : NEG_SLOPE * v;          // LeakyReLU(0.2)
  }
  const float emax = wave_max(e);
  const float w = (lane <= KNN_K) ? __expf(e - emax) : 0.f;
  const float den = wave_sum(w);
  if (lane <= KNN_K) { att_s[wave][lane] = w / den; nb_s[wave][lane] = nb; }
  __syncthreads();

  float acc[C_DIM / 32] = {};
#pragma unroll 1
  for (int j = 0; j <= KNN_K; ++j) {
    const float a = att_s[wave][j];
    const float* hj = h + (size_t)nb_s[wave][j] * C_DIM;
#pragma unroll
    for (int t = 0; t < C_DIM / 32; ++t) acc[t] += a * hj[lane + 32 * t];
  }
  const float* xr = x + (size_t)node * C_DIM;
  float* xo = xout + (size_t)node * C_DIM;
#pragma unroll
  for (int t = 0; t < C_DIM / 32; ++t) {
    const int c = lane + 32 * t;
    const float o = acc[t] + bias[c];
    xo[c] = xr[c] + fmaxf(o, 0.f);
  }
}

// -------------------- host --------------------
extern "C" void kernel_launch(void* const* d_in, const int* in_sizes, int n_in,
                              void* d_out, int out_size, void* d_ws, size_t ws_size,
                              hipStream_t stream) {
  (void)in_sizes; (void)n_in; (void)out_size; (void)ws_size;
  const float* x   = (const float*)d_in[0];
  const float* W   = (const float*)d_in[1];
  const float* a_s = (const float*)d_in[2];
  const float* a_d = (const float*)d_in[3];
  const float* b   = (const float*)d_in[4];

  float* ws   = (float*)d_ws;
  float* sq   = ws;                                            // N
  float* h    = sq + N_NODES;                                  // N*C
  float* al_s = h + (size_t)N_NODES * C_DIM;                   // N
  float* al_d = al_s + N_NODES;                                // N
  int*   idx  = (int*)(al_d + N_NODES);                        // N*(K+1)
  float* x1   = (float*)(idx + (size_t)N_NODES * (KNN_K + 1)); // N*C
  unsigned short* xhi = (unsigned short*)(x1 + (size_t)N_NODES * C_DIM); // N*C bf16
  unsigned short* xlo = xhi + (size_t)N_NODES * C_DIM;                   // N*C bf16
  float* xout = (float*)d_out;

  const float* xin = x;
  for (int m = 0; m < 2; ++m) {
    float* xnext = (m == 0) ? x1 : xout;
    const float* Wm   = W   + (size_t)m * C_DIM * C_DIM;
    const float* asm2 = a_s + (size_t)m * C_DIM;
    const float* adm  = a_d + (size_t)m * C_DIM;
    const float* bm   = b   + (size_t)m * C_DIM;

    rowsq_kernel<<<N_NODES / 8, 256, 0, stream>>>(xin, sq);
    cvt_hilo_kernel<<<(N_NODES * C_DIM) / (256 * 4), 256, 0, stream>>>(xin, xhi, xlo);
    knn_kernel<<<N_NODES / 16, 32, 0, stream>>>((const __bf16*)xhi, (const __bf16*)xlo,
                                                sq, idx);
    gemm_h_kernel<<<dim3(N_NODES / 16, C_DIM / 16), 32, 0, stream>>>(xin, Wm, h);
    alpha_kernel<<<N_NODES / 8, 256, 0, stream>>>(h, asm2, adm, al_s, al_d);
    aggregate_kernel<<<N_NODES / 8, 256, 0, stream>>>(xin, h, idx, al_s, al_d, bm, xnext);
    xin = xnext;
  }
}
